// SquadQALayer_58712202936636
// MI455X (gfx1250) — matrix operand order repaired and checked by reading, not compiled, not run on hardware
//
#include <hip/hip_runtime.h>
#include <hip/hip_bf16.h>
#include <math.h>

typedef __attribute__((ext_vector_type(16))) _Float16 v16h;
typedef __attribute__((ext_vector_type(8)))  _Float16 v8h;
typedef __attribute__((ext_vector_type(8)))  float    v8f;

#define LL 512
#define BB 32
#define HH 1024
#define KK 5
#define RR (LL*BB)
#define MASK_NEG 1e30f

// ---------------- prep: transpose+convert top half of w_end0 to f16 [N,K] ----
__global__ __launch_bounds__(256)
void k_prep_w(const float* __restrict__ w_end0, _Float16* __restrict__ WhT) {
  size_t idx = (size_t)blockIdx.x * 256 + threadIdx.x;   // [0, H*H)
  int c = (int)(idx >> 10), n = (int)(idx & 1023);
  WhT[(size_t)n * HH + c] = (_Float16)w_end0[idx];
}

// ---------------- prep: gw = gamma*w1, Gsum, Bsum ----------------------------
__global__ __launch_bounds__(256)
void k_prep_vec(const float* __restrict__ gamma, const float* __restrict__ beta,
                const float* __restrict__ w1, float* __restrict__ gw,
                float* __restrict__ consts) {
  __shared__ float rg[256], rb[256];
  int t = threadIdx.x;
  float sg = 0.f, sb = 0.f;
  for (int h = t; h < HH; h += 256) {
    float gv = gamma[h] * w1[h];
    gw[h] = gv; sg += gv; sb += beta[h] * w1[h];
  }
  rg[t] = sg; rb[t] = sb; __syncthreads();
  for (int s = 128; s > 0; s >>= 1) {
    if (t < s) { rg[t] += rg[t+s]; rb[t] += rb[t+s]; }
    __syncthreads();
  }
  if (t == 0) { consts[0] = rg[0]; consts[1] = rb[0]; }
}

// ---------------- prep: convert inputs to f16 --------------------------------
__global__ __launch_bounds__(256)
void k_cvt_x(const float* __restrict__ X, _Float16* __restrict__ Xh) {
  size_t base = ((size_t)blockIdx.x * 256 + threadIdx.x) * 8;
  #pragma unroll
  for (int i = 0; i < 8; ++i) Xh[base + i] = (_Float16)X[base + i];
}

// ---------------- start logits (masked), one wave per (l,b) row --------------
__global__ __launch_bounds__(256)
void k_start_logits(const float* __restrict__ X, const float* __restrict__ w_start,
                    const float* __restrict__ b_start, const float* __restrict__ p_mask,
                    float* __restrict__ slm) {
  int wave = threadIdx.x >> 5, ln = threadIdx.x & 31;
  int r = blockIdx.x * 8 + wave;           // r = l*32 + b
  const float* xr = X + (size_t)r * HH;
  float s = 0.f;
  for (int h = ln; h < HH; h += 32) s += xr[h] * w_start[h];
  for (int m = 16; m >= 1; m >>= 1) s += __shfl_xor(s, m, 32);
  if (ln == 0) {
    int l = r >> 5, b = r & 31;
    float v = s + b_start[0];
    float pm = p_mask[b * LL + l];
    slm[b * LL + l] = v * (1.f - pm) - MASK_NEG * pm;
  }
}

// ---------------- start softmax + top-5 --------------------------------------
__global__ __launch_bounds__(512)
void k_start_topk(const float* __restrict__ slm, float* __restrict__ start_p,
                  int* __restrict__ sidx, float* __restrict__ out) {
  __shared__ float sval[512];
  __shared__ int   sind[512];
  __shared__ float red[512];
  int b = blockIdx.x, t = threadIdx.x;
  float v = slm[b * LL + t];
  red[t] = v; __syncthreads();
  for (int s = 256; s > 0; s >>= 1) { if (t < s) red[t] = fmaxf(red[t], red[t+s]); __syncthreads(); }
  float mx = red[0]; __syncthreads();
  float e = expf(v - mx);
  red[t] = e; __syncthreads();
  for (int s = 256; s > 0; s >>= 1) { if (t < s) red[t] += red[t+s]; __syncthreads(); }
  float sum = red[0]; __syncthreads();
  start_p[b * LL + t] = e / sum;
  float lse = mx + logf(sum);
  float myv = v;
  for (int j = 0; j < 5; ++j) {
    sval[t] = myv; sind[t] = t; __syncthreads();
    for (int s = 256; s > 0; s >>= 1) {
      if (t < s) {
        float v2 = sval[t+s]; int i2 = sind[t+s];
        if (v2 > sval[t] || (v2 == sval[t] && i2 < sind[t])) { sval[t] = v2; sind[t] = i2; }
      }
      __syncthreads();
    }
    float wv = sval[0]; int wi = sind[0]; __syncthreads();
    if (t == 0) {
      out[b * 5 + j]       = wv - lse;       // start_top_log_probs
      out[160 + b * 5 + j] = (float)wi;      // start_top_index
      sidx[b * 5 + j] = wi;
    }
    if (t == wi) myv = -3.0e38f;
    __syncthreads();
  }
}

// ---------------- attention-pooled start feature + gather top-k rows ---------
__global__ __launch_bounds__(256)
void k_start_feature(const float* __restrict__ X, const float* __restrict__ start_p,
                     const int* __restrict__ sidx, float* __restrict__ sfw,
                     float* __restrict__ sfeat) {
  __shared__ float p[LL];
  int b = blockIdx.x, t = threadIdx.x;
  for (int l = t; l < LL; l += 256) p[l] = start_p[b * LL + l];
  __syncthreads();
  float acc[4] = {0.f, 0.f, 0.f, 0.f};
  for (int l = 0; l < LL; ++l) {
    const float* xr = X + ((size_t)l * BB + b) * HH;
    float pv = p[l];
    #pragma unroll
    for (int i = 0; i < 4; ++i) acc[i] += pv * xr[t + i * 256];
  }
  #pragma unroll
  for (int i = 0; i < 4; ++i) sfw[b * HH + t + i * 256] = acc[i];
  for (int k = 0; k < 5; ++k) {
    int idx = sidx[b * 5 + k];
    const float* xr = X + ((size_t)idx * BB + b) * HH;
    #pragma unroll
    for (int i = 0; i < 4; ++i) sfeat[(size_t)(b * 5 + k) * HH + t + i * 256] = xr[t + i * 256];
  }
}

// ---------------- sf_proj = start_feat @ w_end0[H:2H] + b_end0 ---------------
__global__ __launch_bounds__(256)
void k_sf_proj(const float* __restrict__ sfeat, const float* __restrict__ w_end0,
               const float* __restrict__ b_end0, float* __restrict__ sfp) {
  int tc = threadIdx.x & 15, tr = threadIdx.x >> 4;
  int col = blockIdx.x * 16 + tc;
  int row = blockIdx.y * 16 + tr;           // row in [0,160)
  const float* sr = sfeat + (size_t)row * HH;
  const float* wc = w_end0 + (size_t)HH * HH + col;   // bottom half, column col
  float acc = 0.f;
  for (int c = 0; c < HH; ++c) acc += sr[c] * wc[(size_t)c * HH];
  sfp[(size_t)row * HH + col] = acc + b_end0[col];
}

// ---------------- main WMMA GEMM: XW = Xh[16384,1024] @ WhT^T ----------------
// 4 waves/WG; wave: M=64 (4 strips) x N=64 (4 tiles); B frags reused over strips.
__global__ __launch_bounds__(128)
void k_gemm(const _Float16* __restrict__ Xh, const _Float16* __restrict__ WhT,
            float* __restrict__ XW) {
  const int wave = threadIdx.x >> 5;
  const int ln   = threadIdx.x & 31;
  const int l15  = ln & 15;
  const int kh   = ln >> 4;
  const int rbase = blockIdx.y * 64;
  const int cbase = blockIdx.x * 256 + wave * 64;

  v8f acc[4][4];
  #pragma unroll
  for (int s = 0; s < 4; ++s)
    #pragma unroll
    for (int t = 0; t < 4; ++t)
      acc[s][t] = v8f{0.f,0.f,0.f,0.f,0.f,0.f,0.f,0.f};

  for (int k0 = 0; k0 < HH; k0 += 32) {
    v16h bfrag[4];
    #pragma unroll
    for (int t = 0; t < 4; ++t) {
      // B 32x16 f16 fragment: lane holds 16 consecutive K at K = kh*16
      const _Float16* bp = WhT + (size_t)(cbase + t * 16 + l15) * HH + k0 + kh * 16;
      v8h blo = *(const v8h*)(bp);
      v8h bhi = *(const v8h*)(bp + 8);
      v16h bf;
      #pragma unroll
      for (int i = 0; i < 8; ++i) { bf[i] = blo[i]; bf[i + 8] = bhi[i]; }
      bfrag[t] = bf;
    }
    #pragma unroll
    for (int s = 0; s < 4; ++s) {
      // A 16x32 f16 fragment: lane holds 16 consecutive K at K = kh*16
      const _Float16* ap = Xh + (size_t)(rbase + s * 16 + l15) * HH + k0 + kh * 16;
      v8h alo = *(const v8h*)(ap);
      v8h ahi = *(const v8h*)(ap + 8);
      v16h af;
      #pragma unroll
      for (int i = 0; i < 8; ++i) { af[i] = alo[i]; af[i + 8] = ahi[i]; }
      #pragma unroll
      for (int t = 0; t < 4; ++t)
        acc[s][t] = __builtin_amdgcn_wmma_f32_16x16x32_f16(
            false, af, false, bfrag[t], (short)0, acc[s][t], false, false);
    }
  }
  // C/D layout: VGPR i -> M = i + 8*(lane>>4), N = lane&15
  #pragma unroll
  for (int s = 0; s < 4; ++s)
    #pragma unroll
    for (int t = 0; t < 4; ++t)
      #pragma unroll
      for (int i = 0; i < 8; ++i)
        XW[(size_t)(rbase + s * 16 + i + 8 * kh) * HH + cbase + t * 16 + l15] = acc[s][t][i];
}

// ---------------- fused tanh + LayerNorm + dot(w_end1) + mask ----------------
__global__ __launch_bounds__(256)
void k_end_ep(const float* __restrict__ XW, const float* __restrict__ sfp,
              const float* __restrict__ gw, const float* __restrict__ consts,
              const float* __restrict__ b_end1, const float* __restrict__ p_mask,
              float* __restrict__ elog) {
  int wave = threadIdx.x >> 5, ln = threadIdx.x & 31;
  int r = blockIdx.x * 8 + wave;
  int l = r >> 5, b = r & 31;
  const float* xr = XW + (size_t)r * HH;
  float s1[5] = {0,0,0,0,0}, s2[5] = {0,0,0,0,0}, s3[5] = {0,0,0,0,0};
  for (int h = ln; h < HH; h += 32) {
    float x = xr[h], gv = gw[h];
    #pragma unroll
    for (int k = 0; k < 5; ++k) {
      float t = tanhf(x + sfp[(size_t)(b * 5 + k) * HH + h]);
      s1[k] += t; s2[k] += t * t; s3[k] += t * gv;
    }
  }
  #pragma unroll
  for (int k = 0; k < 5; ++k)
    for (int m = 16; m >= 1; m >>= 1) {
      s1[k] += __shfl_xor(s1[k], m, 32);
      s2[k] += __shfl_xor(s2[k], m, 32);
      s3[k] += __shfl_xor(s3[k], m, 32);
    }
  if (ln == 0) {
    float Gsum = consts[0], Bsum = consts[1], pm = p_mask[b * LL + l];
    float be1 = b_end1[0];
    #pragma unroll
    for (int k = 0; k < 5; ++k) {
      float mu = s1[k] * (1.f / HH);
      float var = s2[k] * (1.f / HH) - mu * mu;
      float rstd = rsqrtf(var + 1e-12f);
      float lg = rstd * (s3[k] - mu * Gsum) + Bsum + be1;
      elog[(size_t)(b * 5 + k) * LL + l] = lg * (1.f - pm) - MASK_NEG * pm;
    }
  }
}

// ---------------- end log_softmax + top-5 per (b,k) --------------------------
__global__ __launch_bounds__(256)
void k_end_topk(const float* __restrict__ elog, float* __restrict__ out) {
  __shared__ float rv[256]; __shared__ int ri[256];
  int row = blockIdx.x, t = threadIdx.x;   // row = b*5 + k
  float v0 = elog[(size_t)row * LL + t];
  float v1 = elog[(size_t)row * LL + 256 + t];
  rv[t] = fmaxf(v0, v1); __syncthreads();
  for (int s = 128; s > 0; s >>= 1) { if (t < s) rv[t] = fmaxf(rv[t], rv[t+s]); __syncthreads(); }
  float mx = rv[0]; __syncthreads();
  rv[t] = expf(v0 - mx) + expf(v1 - mx); __syncthreads();
  for (int s = 128; s > 0; s >>= 1) { if (t < s) rv[t] += rv[t+s]; __syncthreads(); }
  float lse = mx + logf(rv[0]); __syncthreads();
  int b = row / 5, k = row % 5;
  for (int j = 0; j < 5; ++j) {
    float bv; int bi;
    if (v0 >= v1) { bv = v0; bi = t; } else { bv = v1; bi = 256 + t; }
    rv[t] = bv; ri[t] = bi; __syncthreads();
    for (int s = 128; s > 0; s >>= 1) {
      if (t < s) {
        if (rv[t+s] > rv[t] || (rv[t+s] == rv[t] && ri[t+s] < ri[t])) { rv[t] = rv[t+s]; ri[t] = ri[t+s]; }
      }
      __syncthreads();
    }
    float wv = rv[0]; int wi = ri[0]; __syncthreads();
    if (t == 0) {
      out[320  + b * 25 + k * 5 + j] = wv - lse;   // end_top_log_probs
      out[1120 + b * 25 + k * 5 + j] = (float)wi;  // end_top_index
    }
    if (wi == t) v0 = -3.0e38f;
    if (wi == 256 + t) v1 = -3.0e38f;
    __syncthreads();
  }
}

// ---------------- answerability classifier -----------------------------------
__global__ __launch_bounds__(256)
void k_ans(const float* __restrict__ X, const float* __restrict__ sfw,
           const float* __restrict__ w_ans0, const float* __restrict__ b_ans0,
           const float* __restrict__ w_ans1, float* __restrict__ out) {
  __shared__ float a[2 * HH];
  __shared__ float red[256];
  int b = blockIdx.x, t = threadIdx.x;
  for (int c = t; c < HH; c += 256) {
    a[c] = sfw[b * HH + c];
    a[HH + c] = X[(size_t)b * HH + c];   // inputs[l=0] row: r = b
  }
  __syncthreads();
  float acc[4] = {0.f, 0.f, 0.f, 0.f};
  for (int c = 0; c < 2 * HH; ++c) {
    float av = a[c];
    const float* wr = w_ans0 + (size_t)c * HH + t;
    #pragma unroll
    for (int i = 0; i < 4; ++i) acc[i] += av * wr[i * 256];
  }
  float part = 0.f;
  #pragma unroll
  for (int i = 0; i < 4; ++i) {
    int h = t + i * 256;
    part += tanhf(acc[i] + b_ans0[h]) * w_ans1[h];
  }
  red[t] = part; __syncthreads();
  for (int s = 128; s > 0; s >>= 1) { if (t < s) red[t] += red[t+s]; __syncthreads(); }
  if (t == 0) out[1920 + b] = red[0];
}

extern "C" void kernel_launch(void* const* d_in, const int* in_sizes, int n_in,
                              void* d_out, int out_size, void* d_ws, size_t ws_size,
                              hipStream_t stream) {
  (void)in_sizes; (void)n_in; (void)out_size; (void)ws_size;
  const float* X       = (const float*)d_in[0];
  const float* p_mask  = (const float*)d_in[1];
  const float* w_start = (const float*)d_in[2];
  const float* b_start = (const float*)d_in[3];
  const float* w_end0  = (const float*)d_in[4];
  const float* b_end0  = (const float*)d_in[5];
  const float* g       = (const float*)d_in[6];
  const float* be      = (const float*)d_in[7];
  const float* w_end1  = (const float*)d_in[8];
  const float* b_end1  = (const float*)d_in[9];
  const float* w_ans0  = (const float*)d_in[10];
  const float* b_ans0  = (const float*)d_in[11];
  const float* w_ans1  = (const float*)d_in[12];
  float* out = (float*)d_out;

  char* ws = (char*)d_ws;
  size_t off = 0;
  auto alloc = [&](size_t bytes) -> char* {
    char* p = ws + off; off = (off + bytes + 255) & ~(size_t)255; return p;
  };
  _Float16* WhT  = (_Float16*)alloc((size_t)HH * HH * 2);
  _Float16* Xh   = (_Float16*)alloc((size_t)RR * HH * 2);
  float* XW      = (float*)alloc((size_t)RR * HH * 4);
  float* slm     = (float*)alloc((size_t)BB * LL * 4);
  float* sp      = (float*)alloc((size_t)BB * LL * 4);
  int*   sidx    = (int*)  alloc((size_t)BB * KK * 4);
  float* sfeat   = (float*)alloc((size_t)BB * KK * HH * 4);
  float* sfp     = (float*)alloc((size_t)BB * KK * HH * 4);
  float* sfw     = (float*)alloc((size_t)BB * HH * 4);
  float* gw      = (float*)alloc((size_t)HH * 4);
  float* consts  = (float*)alloc(256);
  float* elog    = (float*)alloc((size_t)BB * KK * LL * 4);

  k_prep_w      <<<4096, 256, 0, stream>>>(w_end0, WhT);
  k_prep_vec    <<<1, 256, 0, stream>>>(g, be, w_end1, gw, consts);
  k_cvt_x       <<<8192, 256, 0, stream>>>(X, Xh);
  k_start_logits<<<2048, 256, 0, stream>>>(X, w_start, b_start, p_mask, slm);
  k_start_topk  <<<32, 512, 0, stream>>>(slm, sp, sidx, out);
  k_start_feature<<<32, 256, 0, stream>>>(X, sp, sidx, sfw, sfeat);
  k_sf_proj     <<<dim3(64, 10), 256, 0, stream>>>(sfeat, w_end0, b_end0, sfp);
  k_gemm        <<<dim3(4, 256), 128, 0, stream>>>(Xh, WhT, XW);
  k_end_ep      <<<2048, 256, 0, stream>>>(XW, sfp, gw, consts, b_end1, p_mask, elog);
  k_end_topk    <<<160, 256, 0, stream>>>(elog, out);
  k_ans         <<<32, 256, 0, stream>>>(X, sfw, w_ans0, b_ans0, w_ans1, out);
}
